// ProtoMemory_35296041238691
// MI455X (gfx1250) — matrix operand order repaired and checked by reading, not compiled, not run on hardware
//
#include <hip/hip_runtime.h>
#include <hip/hip_bf16.h>

// ---------------------------------------------------------------------------
// Problem constants (from reference setup_inputs):
//   B=8, C=256, H=W=64, HW=4096, feat=F=128, K=128, P=256, total=32768,
//   hidden=256, M_ALL = B*HW = 32768.
// All GEMMs here have B/C leading dimension = HW and lda = KDIM (dense A),
// which we exploit as compile-time constants so every strided access lowers
// to immediate-offset vector memory ops (no serialized address chains).
// ---------------------------------------------------------------------------
#define BB 8
#define CC 256
#define HW 4096
#define FF 128
#define KK 128
#define PP 256
#define TOT 32768
#define HID 256
#define M_ALL (BB*HW)

typedef __attribute__((ext_vector_type(16))) __bf16 v16bf;
typedef __attribute__((ext_vector_type(8)))  float  v8f;

struct GemmP {
  const float* A; long long sA;   // per-batch element stride (0 for shared weights)
  const float* Bm; long long sB;
  float*       C; long long sC;
  const float* bias;                       // EPI 1,2 (per-row)
  const float* add;  long long sAdd;       // EPI 3 (same shape as C)
  const float* epx;  long long sX;         // EPI 4 residual input
  const float* scale;                      // EPI 4 per-row gamma*rsqrt(var+eps)
  const float* shift;                      // EPI 4 per-row beta - scale*mean
};

// ---------------------------------------------------------------------------
// 128x64-tile GEMM, C = A(MxK,row,lda=KDIM) * B(KxN,row,ldb=HW), fp32 in/out,
// bf16 WMMA core. Block = 128 threads = 4 waves; wave w owns rows
// [32w,32w+32) (2 A fragments, 8 accumulators). B tile is staged into LDS in
// contiguous-k fragment order, k = 16*(lane/16) + e (4 subtiles x 32 lanes x 16 halves contiguous per
// lane) so each fragment read is one aligned 32B LDS vector load; the gather
// uses compile-time strides so it lowers to an immediate-offset load clause.
// ---------------------------------------------------------------------------
template<int EPI, int KDIM>
__global__ __launch_bounds__(128)
void gemm_bf16_wmma(GemmP p) {
  const int bId = blockIdx.z;
  const float* A  = p.A  + (long long)bId * p.sA;
  const float* Bg = p.Bm + (long long)bId * p.sB;
  float*       C  = p.C  + (long long)bId * p.sC;

  const int n0 = blockIdx.x * 64;
  const int m0 = blockIdx.y * 128;

  __shared__ alignas(32) __bf16 BsF[4 * 32 * 16];   // fragment-packed B tile (4KB)

  const int tid  = threadIdx.x;
  const int wave = tid >> 5;
  const int lane = tid & 31;
  const int lrow = lane & 15;
  const int lhi  = lane >> 4;

  // B staging: thread tid owns fragment slot tid = (subtile<<5)|slotlane
  const int sn = n0 + (tid >> 5) * 16 + (tid & 15);
  const int sk = ((tid >> 4) & 1) * 16;

  v8f acc[2][4];
#pragma unroll
  for (int h = 0; h < 2; ++h)
#pragma unroll
    for (int t = 0; t < 4; ++t)
#pragma unroll
      for (int e = 0; e < 8; ++e) acc[h][t][e] = 0.0f;

#pragma unroll
  for (int kk = 0; kk < KDIM / 32; ++kk) {
    const int k0 = kk * 32;

    // ---- stage B tile (32 K x 64 N) in fragment order ----
    {
      const float* bp = Bg + (long long)(k0 + sk) * HW + sn;  // HW constexpr
      v16bf pk;
#pragma unroll
      for (int e = 0; e < 16; ++e)
        pk[e] = (__bf16)bp[e * HW];          // immediate offsets e*16384B
      *(v16bf*)(BsF + (tid << 4)) = pk;      // 32B aligned vector store
    }

    // ---- A fragments direct from global (row-major, 16B-aligned float4s) ----
    v16bf af[2];
#pragma unroll
    for (int h = 0; h < 2; ++h) {
      const float* ar =
          A + (long long)(m0 + wave * 32 + h * 16 + lrow) * KDIM + k0;
      const float4* a0 = (const float4*)(ar + lhi * 16);
      const float4* a1 = (const float4*)(ar + 8 + lhi * 16);
      float4 x0 = a0[0], x1 = a0[1];
      float4 y0 = a1[0], y1 = a1[1];
      af[h][0]  = (__bf16)x0.x; af[h][1]  = (__bf16)x0.y;
      af[h][2]  = (__bf16)x0.z; af[h][3]  = (__bf16)x0.w;
      af[h][4]  = (__bf16)x1.x; af[h][5]  = (__bf16)x1.y;
      af[h][6]  = (__bf16)x1.z; af[h][7]  = (__bf16)x1.w;
      af[h][8]  = (__bf16)y0.x; af[h][9]  = (__bf16)y0.y;
      af[h][10] = (__bf16)y0.z; af[h][11] = (__bf16)y0.w;
      af[h][12] = (__bf16)y1.x; af[h][13] = (__bf16)y1.y;
      af[h][14] = (__bf16)y1.z; af[h][15] = (__bf16)y1.w;
    }

    __syncthreads();

    // ---- 4 B fragments x 2 row-halves of WMMA ----
#pragma unroll
    for (int t = 0; t < 4; ++t) {
      v16bf bf = *(const v16bf*)(BsF + ((t * 32 + lane) << 4));
#pragma unroll
      for (int h = 0; h < 2; ++h)
        acc[h][t] = __builtin_amdgcn_wmma_f32_16x16x32_bf16(
            false, af[h], false, bf, (short)0, acc[h][t], false, false);
    }
    __syncthreads();
  }

  // ---- epilogue + store: VGPR r -> M = r + 8*(lane>=16), N = lane%16 ----
#pragma unroll
  for (int h = 0; h < 2; ++h) {
#pragma unroll
    for (int t = 0; t < 4; ++t) {
#pragma unroll
      for (int r = 0; r < 8; ++r) {
        const int row = m0 + wave * 32 + h * 16 + r + 8 * lhi;
        const int col = n0 + t * 16 + lrow;
        const long long off = (long long)row * HW + col;
        float v = acc[h][t][r];
        if constexpr (EPI == 1) { v += p.bias[row]; v = fmaxf(v, 0.0f); }
        if constexpr (EPI == 2) { v += p.bias[row]; }
        if constexpr (EPI == 3) { v += (p.add + (long long)bId * p.sAdd)[off]; }
        if constexpr (EPI == 4) {
          const float xz = (p.epx + (long long)bId * p.sX)[off] + v;
          v = p.scale[row] * xz + p.shift[row];
        }
        C[off] = v;
      }
    }
  }
}

// ---------------------------------------------------------------------------
// Helper kernels
// ---------------------------------------------------------------------------
__global__ void transpose128(const float* __restrict__ src, float* __restrict__ dst) {
  int i = blockIdx.x * blockDim.x + threadIdx.x;      // 0..16383
  if (i >= FF * KK) return;
  int f = i >> 7, k = i & 127;
  dst[k * FF + f] = src[f * KK + k];
}

__global__ void bn_prep(const float* g, const float* b, const float* m,
                        const float* v, float* scale, float* shift) {
  int i = blockIdx.x * blockDim.x + threadIdx.x;
  if (i >= CC) return;
  float s = g[i] * rsqrtf(v[i] + 1e-5f);
  scale[i] = s;
  shift[i] = b[i] - s * m[i];
}

// softmax over K (major axis), hard_shrink_relu, renormalize — in place on [B][K][HW]
__global__ void softmax_shrink(float* logit) {
  int idx = blockIdx.x * blockDim.x + threadIdx.x;
  if (idx >= M_ALL) return;
  int b = idx / HW, m = idx % HW;
  float* p = logit + (long long)b * (KK * HW) + m;
  float mx = -3.4e38f;
  for (int k = 0; k < KK; ++k) mx = fmaxf(mx, p[k * HW]);
  float sum = 0.f;
  for (int k = 0; k < KK; ++k) {
    float e = __expf(p[k * HW] - mx);
    p[k * HW] = e; sum += e;
  }
  const float inv = 1.f / sum;
  const float lambd = 1.f / (float)KK;
  float s2 = 0.f;
  for (int k = 0; k < KK; ++k) {
    float a = p[k * HW] * inv;
    float s = a - lambd;
    float a2 = fmaxf(s, 0.f) * a / (fabsf(s) + 1e-12f);
    p[k * HW] = a2; s2 += a2;
  }
  const float inv2 = 1.f / (s2 + 1e-12f);
  for (int k = 0; k < KK; ++k) p[k * HW] *= inv2;
}

// argmax over K of scores[B][K][HW] -> pool column index
__global__ void argmax_idx(const float* __restrict__ scores,
                           const int* __restrict__ rand_slot, int* __restrict__ idx) {
  int m = blockIdx.x * blockDim.x + threadIdx.x;
  if (m >= M_ALL) return;
  int b = m / HW, hw = m % HW;
  const float* p = scores + (long long)b * (KK * HW) + hw;
  float best = p[0]; int bi = 0;
  for (int k = 1; k < KK; ++k) {
    float v = p[k * HW];
    if (v > best) { best = v; bi = k; }
  }
  idx[m] = bi * PP + rand_slot[m];
}

__global__ void copyf(const float* __restrict__ s, float* __restrict__ d, int n) {
  for (int i = blockIdx.x * blockDim.x + threadIdx.x; i < n;
       i += gridDim.x * blockDim.x) d[i] = s[i];
}

__global__ void fillneg(int* w, int n) {
  int i = blockIdx.x * blockDim.x + threadIdx.x;
  if (i < n) w[i] = -1;
}

// deterministic duplicate resolution: largest m writing a column wins
__global__ void vote(const int* __restrict__ idx, int* __restrict__ winner) {
  int m = blockIdx.x * blockDim.x + threadIdx.x;
  if (m >= M_ALL) return;
  atomicMax(&winner[idx[m]], m);
}

// pool[:, idx[m]] = act[b][:, hw] for winning m; act is [B][F][HW]
__global__ void scatter_pool(const float* __restrict__ act, const int* __restrict__ idx,
                             const int* __restrict__ winner, float* __restrict__ pool) {
  int m = blockIdx.x;
  int f = threadIdx.x;                 // 0..127
  int col = idx[m];
  if (winner[col] != m) return;
  int b = m / HW, hw = m % HW;
  pool[(long long)f * TOT + col] =
      act[(long long)b * (FF * HW) + f * HW + hw];
}

// ---------------------------------------------------------------------------
extern "C" void kernel_launch(void* const* d_in, const int* in_sizes, int n_in,
                              void* d_out, int out_size, void* d_ws, size_t ws_size,
                              hipStream_t stream) {
  (void)in_sizes; (void)n_in; (void)out_size; (void)ws_size;
  const float* x        = (const float*)d_in[0];
  const float* theta_w  = (const float*)d_in[1];
  const float* phi_k_w  = (const float*)d_in[2];
  const float* g_w      = (const float*)d_in[3];
  const float* o_w      = (const float*)d_in[4];
  const float* fc1_w    = (const float*)d_in[5];
  const float* fc1_b    = (const float*)d_in[6];
  const float* fc2_w    = (const float*)d_in[7];
  const float* fc2_b    = (const float*)d_in[8];
  const float* cpool    = (const float*)d_in[9];
  const float* cproto   = (const float*)d_in[10];
  const float* bn_gamma = (const float*)d_in[11];
  const float* bn_beta  = (const float*)d_in[12];
  const float* bn_mean  = (const float*)d_in[13];
  const float* bn_var   = (const float*)d_in[14];
  const int*   rslot    = (const int*)d_in[15];

  float* y_out    = (float*)d_out;                       // [B][C][HW]
  float* pool_out = y_out + (long long)BB * CC * HW;     // [F][TOT]

  // ---- workspace carve-up (floats) ----
  float* ws = (float*)d_ws;
  float* protoT  = ws;                 ws += FF * KK;        // [K][F]
  float* bnscale = ws;                 ws += CC;
  float* bnshift = ws;                 ws += CC;
  float* thetaB  = ws;                 ws += (long long)BB * FF  * HW;  // reused as read+g
  float* gB      = ws;                 ws += (long long)BB * FF  * HW;
  float* actB    = ws;                 ws += (long long)BB * FF  * HW;
  float* energyB = ws;                 ws += (long long)BB * KK  * HW;  // reused as scores
  float* hidB    = ws;                 ws += (long long)BB * HID * HW;
  float* logitB  = ws;                 ws += (long long)BB * KK  * HW;
  int* idxB      = (int*)ws;
  int* winnerB   = idxB + M_ALL;

  const long long sX   = (long long)CC * HW;     // x / y batch stride
  const long long sF   = (long long)FF * HW;     // 128-row buffers
  const long long sHid = (long long)HID * HW;

  // prep
  transpose128<<<64, 256, 0, stream>>>(cproto, protoT);
  bn_prep<<<1, 256, 0, stream>>>(bn_gamma, bn_beta, bn_mean, bn_var, bnscale, bnshift);

  const dim3 blk(128);
  const dim3 g128(HW / 64, FF / 128, BB);     // M=128 outputs (gy=1)
  const dim3 g256(HW / 64, CC / 128, BB);     // M=256 outputs (gy=2)

  GemmP p{};

  // 1) theta_x = theta_w @ X_b            [F x HW]
  p.A = theta_w; p.sA = 0;
  p.Bm = x; p.sB = sX; p.C = thetaB; p.sC = sF;
  gemm_bf16_wmma<0, CC><<<g128, blk, 0, stream>>>(p);
  // 2) g_x = g_w @ X_b
  p.A = g_w; p.C = gB;
  gemm_bf16_wmma<0, CC><<<g128, blk, 0, stream>>>(p);
  // 3) act = phi_k_w @ X_b
  p.A = phi_k_w; p.C = actB;
  gemm_bf16_wmma<0, CC><<<g128, blk, 0, stream>>>(p);
  // 4) energyT = protoT @ theta_x         [K x HW]
  p.A = protoT;
  p.Bm = thetaB; p.sB = sF; p.C = energyB; p.sC = sF;
  gemm_bf16_wmma<0, FF><<<g128, blk, 0, stream>>>(p);
  // 5) hidT = relu(fc1_w @ energyT + b)   [HID x HW]
  p.A = fc1_w;
  p.Bm = energyB; p.C = hidB; p.sC = sHid; p.bias = fc1_b;
  gemm_bf16_wmma<1, KK><<<g256, blk, 0, stream>>>(p);
  // 6) logitsT = fc2_w @ hidT + b         [K x HW]
  p.A = fc2_w;
  p.Bm = hidB; p.sB = sHid; p.C = logitB; p.sC = sF; p.bias = fc2_b;
  gemm_bf16_wmma<2, HID><<<g128, blk, 0, stream>>>(p);
  // 7) softmax + hard-shrink + renorm (in place -> attnT)
  softmax_shrink<<<(M_ALL + 255) / 256, 256, 0, stream>>>(logitB);
  // 8) read+g = proto @ attnT + g_x       [F x HW]  (reuse thetaB)
  p.A = cproto;
  p.Bm = logitB; p.sB = sF; p.C = thetaB; p.sC = sF;
  p.add = gB; p.sAdd = sF;
  gemm_bf16_wmma<3, KK><<<g128, blk, 0, stream>>>(p);
  // 9) y = BN(x + o_w @ (read+g))         [C x HW] -> d_out
  p.A = o_w;
  p.Bm = thetaB; p.sB = sF; p.C = y_out; p.sC = sX;
  p.epx = x; p.sX = sX; p.scale = bnscale; p.shift = bnshift;
  gemm_bf16_wmma<4, FF><<<g256, blk, 0, stream>>>(p);
  // 10) scoresT = protoT @ act            [K x HW]  (reuse energyB)
  p.A = protoT;
  p.Bm = actB; p.sB = sF; p.C = energyB; p.sC = sF;
  gemm_bf16_wmma<0, FF><<<g128, blk, 0, stream>>>(p);
  // 11) cluster argmax -> pool column index
  argmax_idx<<<(M_ALL + 255) / 256, 256, 0, stream>>>(energyB, rslot, idxB);
  // 12) new_pool = concept_pool, then deterministic scatter of act columns
  copyf<<<4096, 256, 0, stream>>>(cpool, pool_out, FF * TOT);
  fillneg<<<(TOT + 255) / 256, 256, 0, stream>>>(winnerB, TOT);
  vote<<<(M_ALL + 255) / 256, 256, 0, stream>>>(idxB, winnerB);
  scatter_pool<<<M_ALL, FF, 0, stream>>>(actB, idxB, winnerB, pool_out);
}
